// Transformer_64888365908312
// MI455X (gfx1250) — matrix operand order repaired and checked by reading, not compiled, not run on hardware
//
#include <hip/hip_runtime.h>
#include <hip/hip_bf16.h>

typedef _Float16 f16;
typedef _Float16 v16h __attribute__((ext_vector_type(16)));
typedef float    v8f  __attribute__((ext_vector_type(8)));
typedef unsigned int u32x4 __attribute__((ext_vector_type(4)));
typedef unsigned int u32x8 __attribute__((ext_vector_type(8)));

union Frag16 { v16h v; float4 q[2]; f16 h[16]; };

#define DIMW   1024
#define SEQW   1024
#define BATCHW 8

static __device__ __forceinline__ v8f wmma_f16(const v16h& a, const v16h& b, const v8f& c) {
    return __builtin_amdgcn_wmma_f32_16x16x32_f16(false, a, false, b, (short)0, c, false, false);
}

// ---- DPP row (16-lane) reductions: quad_perm xor1, xor2, row_ror:4, row_ror:8 ----
template<int CTRL>
static __device__ __forceinline__ float dpp_f32(float x) {
    return __builtin_bit_cast(float,
        __builtin_amdgcn_update_dpp(0, __builtin_bit_cast(int, x), CTRL, 0xF, 0xF, true));
}
static __device__ __forceinline__ float rowmax16(float x) {
    x = fmaxf(x, dpp_f32<0xB1>(x));    // quad_perm [1,0,3,2]  (xor 1)
    x = fmaxf(x, dpp_f32<0x4E>(x));    // quad_perm [2,3,0,1]  (xor 2)
    x = fmaxf(x, dpp_f32<0x124>(x));   // row_ror:4
    x = fmaxf(x, dpp_f32<0x128>(x));   // row_ror:8
    return x;
}
static __device__ __forceinline__ float rowsum16(float x) {
    x += dpp_f32<0xB1>(x);
    x += dpp_f32<0x4E>(x);
    x += dpp_f32<0x124>(x);
    x += dpp_f32<0x128>(x);
    return x;
}

// ---- LDS byte address: flat LDS aperture maps addr[31:0] to LDS offset (ISA 10.2) ----
static __device__ __forceinline__ unsigned lds_lo32(const void* p) {
    return (unsigned)(unsigned long long)p;
}

// ---- Tensor Data Mover: 2D tile (f16) global -> LDS with optional row padding ----
// D# packing per CDNA5 ISA. tensor_dim := tile_dim
// (tiles are always fully in-bounds), stride0 = global row pitch in elements.
static __device__ __forceinline__ void tdm_load_2d_f16(
        const void* gsrc, unsigned ldsaddr,
        unsigned tile_d0, unsigned tile_d1, unsigned long long stride0,
        unsigned pad_en, unsigned pad_interval_code, unsigned pad_amount_code) {
    const unsigned long long ga = (unsigned long long)gsrc;
    u32x4 g0;
    g0[0] = 1u;                                         // count=1 (valid), user mode
    g0[1] = ldsaddr;                                    // lds_addr (bytes)
    g0[2] = (unsigned)ga;                               // global_addr[31:0]
    g0[3] = (unsigned)((ga >> 32) & 0x01FFFFFFull)      // global_addr[56:32]
          | 0x80000000u;                                // type=2 ("image")
    u32x8 g1;
    g1[0] = (1u << 16)                                  // data_size=1 (2 bytes)
          | (pad_en << 20)
          | (pad_interval_code << 22)
          | (pad_amount_code << 25);
    g1[1] = (tile_d0 & 0xFFFFu) << 16;                  // tensor_dim0[15:0] @bits63:48
    g1[2] = (tile_d0 >> 16) | ((tile_d1 & 0xFFFFu) << 16); // dim0[31:16] | dim1[15:0]
    g1[3] = (tile_d1 >> 16) | ((tile_d0 & 0xFFFFu) << 16); // dim1[31:16] | tile_dim0
    g1[4] = (tile_d1 & 0xFFFFu);                        // tile_dim1 ; tile_dim2=0
    g1[5] = (unsigned)stride0;                          // tensor_dim0_stride[31:0]
    g1[6] = (unsigned)((stride0 >> 32) & 0xFFFFull);    // stride[47:32] ; stride1=0
    g1[7] = 0u;
    asm volatile("tensor_load_to_lds %0, %1" :: "s"(g0), "s"(g1) : "memory");
}

// ---------------- LayerNorm (row = 1024 fp32) -> f16 ----------------
__global__ __launch_bounds__(128)
void ln_kernel(const float* __restrict__ x, const float* __restrict__ gamma,
               const float* __restrict__ beta, f16* __restrict__ out) {
    __shared__ float rs[128], rq[128];
    const long row = blockIdx.x;
    const int tid = threadIdx.x;
    const float4* xr = (const float4*)(x + row * DIMW);
    float4 a = xr[tid * 2 + 0];
    float4 b = xr[tid * 2 + 1];
    float vals[8] = {a.x, a.y, a.z, a.w, b.x, b.y, b.z, b.w};
    float s = 0.f, ss = 0.f;
#pragma unroll
    for (int j = 0; j < 8; ++j) { s += vals[j]; ss += vals[j] * vals[j]; }
    rs[tid] = s; rq[tid] = ss;
    __syncthreads();
    for (int st = 64; st > 0; st >>= 1) {
        if (tid < st) { rs[tid] += rs[tid + st]; rq[tid] += rq[tid + st]; }
        __syncthreads();
    }
    const float mu   = rs[0] * (1.0f / DIMW);
    const float var  = rq[0] * (1.0f / DIMW) - mu * mu;
    const float rstd = rsqrtf(var + 1e-3f);
    const int c0 = tid * 8;
    union { f16 h[8]; float4 q; } o;
#pragma unroll
    for (int j = 0; j < 8; ++j)
        o.h[j] = (f16)((vals[j] - mu) * rstd * gamma[c0 + j] + beta[c0 + j]);
    *(float4*)(out + row * DIMW + c0) = o.q;
}

// ---------------- weight transpose+cast: in [K][N] fp32 -> out [N][K] f16 ----------------
// Output rows n < scale_rows are multiplied by `scale` (folds the attention 1/sqrt(hd)
// into the Q columns of w_qkv -> zero scale instructions in the flash inner loop).
__global__ __launch_bounds__(256)
void transpose_cast_kernel(const float* __restrict__ in, f16* __restrict__ out,
                           int K, int N, int scale_rows, float scale) {
    __shared__ float tile[32][33];
    const int tx = threadIdx.x & 31;
    const int ty = threadIdx.x >> 5;
    const long k0 = (long)blockIdx.y * 32;
    const long n0 = (long)blockIdx.x * 32;
#pragma unroll
    for (int j = 0; j < 4; ++j)
        tile[ty + j * 8][tx] = in[(k0 + ty + j * 8) * (long)N + n0 + tx];
    __syncthreads();
#pragma unroll
    for (int j = 0; j < 4; ++j) {
        const long n = n0 + ty + j * 8;
        float v = tile[tx][ty + j * 8];
        if (n < scale_rows) v *= scale;
        out[n * (long)K + k0 + tx] = (f16)v;
    }
}

// ---------------- WMMA GEMM: C[M,N] = A[M,K](f16) x BT[N,K](f16) ----------------
// 256 threads = 8 waves (4x2). Block tile 128x128, K-step 32. Wave tile 32x64.
// Both tiles DMA'd by the Tensor Data Mover into padded [128][40] LDS, double buffered.
template<bool OUTF16, bool HASBIAS, bool GELU, bool RESID>
__global__ __launch_bounds__(256, 2)
void gemm_kernel(const f16* __restrict__ A, const f16* __restrict__ BT,
                 const float* __restrict__ bias, const float* __restrict__ resid,
                 void* __restrict__ outp, int M, int N, int K) {
    __shared__ f16 As[2][128][40];   // [m][k] padded: TDM pad 16dw interval, 4dw amount
    __shared__ f16 Bs[2][128][40];   // [n][k] padded
    const int tid  = threadIdx.x;
    const int lane = tid & 31;
    const int wv   = __builtin_amdgcn_readfirstlane(tid >> 5);  // scalar wave id
    const int half_ = lane >> 4, ln16 = lane & 15;
    const int wm = wv & 3, wn = wv >> 2;
    const long row0 = (long)blockIdx.y * 128;
    const long col0 = (long)blockIdx.x * 128;

    v8f acc[2][4] = {};

    // preload tile 0 (wave0: A, wave1: B)
    if (wv == 0)
        tdm_load_2d_f16(A + row0 * (long)K, lds_lo32(&As[0][0][0]), 32, 128,
                        (unsigned long long)K, 1, 3, 3);
    if (wv == 1)
        tdm_load_2d_f16(BT + col0 * (long)K, lds_lo32(&Bs[0][0][0]), 32, 128,
                        (unsigned long long)K, 1, 3, 3);
    if (wv < 2) asm volatile("s_wait_tensorcnt 0x0" ::: "memory");
    __syncthreads();

    for (int kk = 0; kk < K; kk += 32) {
        const int buf = (kk >> 5) & 1;
        if (kk + 32 < K) {   // issue DMA for next tile into the other buffer
            if (wv == 0)
                tdm_load_2d_f16(A + row0 * (long)K + kk + 32,
                                lds_lo32(&As[buf ^ 1][0][0]), 32, 128,
                                (unsigned long long)K, 1, 3, 3);
            if (wv == 1)
                tdm_load_2d_f16(BT + col0 * (long)K + kk + 32,
                                lds_lo32(&Bs[buf ^ 1][0][0]), 32, 128,
                                (unsigned long long)K, 1, 3, 3);
        }

        Frag16 af[2], bf[4];
#pragma unroll
        for (int fm = 0; fm < 2; ++fm) {
            const int m = wm * 32 + fm * 16 + ln16;
            af[fm].q[0] = *(const float4*)&As[buf][m][8 * half_];       // K(e)=e+8*half
            af[fm].q[1] = *(const float4*)&As[buf][m][16 + 8 * half_];  // K(e)=e+8+8*half
        }
#pragma unroll
        for (int fn = 0; fn < 4; ++fn) {
            const int n = wn * 64 + fn * 16 + ln16;
            bf[fn].q[0] = *(const float4*)&Bs[buf][n][8 * half_];       // K(e)=e+8*half
            bf[fn].q[1] = *(const float4*)&Bs[buf][n][16 + 8 * half_];
        }
#pragma unroll
        for (int fm = 0; fm < 2; ++fm)
#pragma unroll
            for (int fn = 0; fn < 4; ++fn)
                acc[fm][fn] = wmma_f16(af[fm].v, bf[fn].v, acc[fm][fn]);

        if (wv < 2 && kk + 32 < K) asm volatile("s_wait_tensorcnt 0x0" ::: "memory");
        __syncthreads();
    }

#pragma unroll
    for (int fm = 0; fm < 2; ++fm) {
#pragma unroll
        for (int fn = 0; fn < 4; ++fn) {
            const long col = col0 + wn * 64 + fn * 16 + ln16;
            const float bv = HASBIAS ? bias[col] : 0.0f;
#pragma unroll
            for (int r = 0; r < 8; ++r) {
                const long row = row0 + wm * 32 + fm * 16 + r + 8 * half_;
                float v = acc[fm][fn][r] + bv;
                if (GELU) {
                    const float u = v;
                    v = 0.5f * u * (1.0f + tanhf(0.7978845608f * (u + 0.044715f * u * u * u)));
                }
                if (RESID) v += resid[row * (long)N + col];
                if (OUTF16) ((f16*)outp)[row * (long)N + col] = (f16)v;
                else        ((float*)outp)[row * (long)N + col] = v;
            }
        }
    }
}

// ---------------- Flash attention: qkv f16 [B*S, 3072] -> o f16 [B*S, 1024] ----------------
// Q columns of w_qkv were pre-scaled by 1/sqrt(hd), so scores come out of WMMA ready.
__global__ __launch_bounds__(256, 2)
void attn_kernel(const f16* __restrict__ qkv, f16* __restrict__ o) {
    __shared__ f16 Kl[32][72];       // [key][hd], TDM pad: 32dw interval, 4dw amount
    __shared__ f16 Vt[64][40];       // transposed V: [hd][key]
    __shared__ f16 Pst[8][16][40];   // per-wave P staging: [wave][m][key]

    const int tid = threadIdx.x, lane = tid & 31;
    const int wv  = __builtin_amdgcn_readfirstlane(tid >> 5);
    const int half_ = lane >> 4, ln16 = lane & 15;
    const int bhq = blockIdx.x;
    const int qt = bhq & 7, h = (bhq >> 3) & 15, b = bhq >> 7;
    const long rowbase = (long)b * SEQW;
    const int q0 = qt * 128 + wv * 16;

    // Q fragments (A-layout), resident for whole kernel
    Frag16 qf[2];
    {
        const long qrow = rowbase + q0 + ln16;
        const f16* qp = qkv + qrow * 3072 + h * 64;
#pragma unroll
        for (int c = 0; c < 2; ++c) {
            qf[c].q[0] = *(const float4*)(qp + c * 32 + 8 * half_);
            qf[c].q[1] = *(const float4*)(qp + c * 32 + 16 + 8 * half_);
        }
    }

    v8f accO[4] = {};
    float mrow[8], lrow[8];
#pragma unroll
    for (int r = 0; r < 8; ++r) { mrow[r] = -1e30f; lrow[r] = 0.0f; }

    const int kkey = tid >> 3, kh = (tid & 7) * 8;
    for (int kb = 0; kb < SEQW / 32; ++kb) {
        __syncthreads();   // previous compute done before tiles are overwritten
        // K tile via Tensor Data Mover (32 keys x 64 dims, padded rows)
        if (wv == 0)
            tdm_load_2d_f16(qkv + (rowbase + kb * 32) * 3072 + 1024 + h * 64,
                            lds_lo32(&Kl[0][0]), 64, 32, 3072ull, 1, 4, 3);
        {   // V tile: manual load + in-LDS transpose
            const long krow = rowbase + kb * 32 + kkey;
            const f16* vp = qkv + krow * 3072 + 2048 + h * 64 + kh;
            float4 vq = *(const float4*)vp;
            __builtin_prefetch(vp + 32 * 3072, 0, 0);   // next kb tile
            union { float4 q; f16 hh[8]; } vu; vu.q = vq;
#pragma unroll
            for (int j = 0; j < 8; ++j) Vt[kh + j][kkey] = vu.hh[j];
        }
        if (wv == 0) asm volatile("s_wait_tensorcnt 0x0" ::: "memory");
        __syncthreads();

        // S = Q K^T over 32 keys -> two 16x16 fragments (already scaled)
        v8f sf[2];
#pragma unroll
        for (int kb2 = 0; kb2 < 2; ++kb2) {
            const int key_l = kb2 * 16 + ln16;
            Frag16 b0, b1;
            b0.q[0] = *(const float4*)&Kl[key_l][8 * half_];
            b0.q[1] = *(const float4*)&Kl[key_l][16 + 8 * half_];
            b1.q[0] = *(const float4*)&Kl[key_l][32 + 8 * half_];
            b1.q[1] = *(const float4*)&Kl[key_l][48 + 8 * half_];
            v8f s = {};
            s = wmma_f16(qf[0].v, b0.v, s);
            s = wmma_f16(qf[1].v, b1.v, s);
            sf[kb2] = s;
        }

        // online softmax: DPP row reductions (16-lane rows), no LDS traffic
#pragma unroll
        for (int r = 0; r < 8; ++r) {
            const float s0 = sf[0][r], s1 = sf[1][r];
            const float mx = rowmax16(fmaxf(s0, s1));
            const float mn = fmaxf(mrow[r], mx);
            const float alpha = __expf(mrow[r] - mn);
            const float p0 = __expf(s0 - mn), p1 = __expf(s1 - mn);
            const float ps = rowsum16(p0 + p1);
            lrow[r] = lrow[r] * alpha + ps;
            mrow[r] = mn;
#pragma unroll
            for (int c = 0; c < 4; ++c) accO[c][r] *= alpha;
            const int mi = r + 8 * half_;
            Pst[wv][mi][ln16]      = (f16)p0;
            Pst[wv][mi][16 + ln16] = (f16)p1;
        }
        // same-wave LDS RAW: DS ops in-order per wave; fence compiler + counter
        asm volatile("s_wait_dscnt 0x0" ::: "memory");

        // P (A-layout) x V -> accumulate O over 4 hd-chunks
        Frag16 pf;
        pf.q[0] = *(const float4*)&Pst[wv][ln16][8 * half_];
        pf.q[1] = *(const float4*)&Pst[wv][ln16][16 + 8 * half_];
#pragma unroll
        for (int c = 0; c < 4; ++c) {
            const int hd = c * 16 + ln16;
            Frag16 vf;
            vf.q[0] = *(const float4*)&Vt[hd][8 * half_];
            vf.q[1] = *(const float4*)&Vt[hd][16 + 8 * half_];
            accO[c] = wmma_f16(pf.v, vf.v, accO[c]);
        }
    }

#pragma unroll
    for (int r = 0; r < 8; ++r) {
        const float inv = 1.0f / lrow[r];
        const long qrow = rowbase + q0 + r + 8 * half_;
        f16* op = o + qrow * DIMW + h * 64;
#pragma unroll
        for (int c = 0; c < 4; ++c) op[c * 16 + ln16] = (f16)(accO[c][r] * inv);
    }
}

// ---------------- host launcher ----------------
extern "C" void kernel_launch(void* const* d_in, const int* in_sizes, int n_in,
                              void* d_out, int out_size, void* d_ws, size_t ws_size,
                              hipStream_t stream) {
    (void)in_sizes; (void)n_in; (void)out_size; (void)ws_size;
    const float* x      = (const float*)d_in[0];
    const float* g1     = (const float*)d_in[1];
    const float* be1    = (const float*)d_in[2];
    const float* w_qkv  = (const float*)d_in[3];
    const float* w_proj = (const float*)d_in[4];
    const float* b_proj = (const float*)d_in[5];
    const float* g2     = (const float*)d_in[6];
    const float* be2    = (const float*)d_in[7];
    const float* w1     = (const float*)d_in[8];
    const float* b1     = (const float*)d_in[9];
    const float* w2     = (const float*)d_in[10];
    const float* b2     = (const float*)d_in[11];
    float* out = (float*)d_out;

    char* ws = (char*)d_ws;
    size_t off = 0;
    auto alloc = [&](size_t bytes) -> void* {
        void* p = ws + off;
        off += (bytes + 255) & ~(size_t)255;
        return p;
    };
    const long M = (long)BATCHW * SEQW;                        // 8192
    f16* wqkvT  = (f16*)alloc((size_t)3072 * 1024 * 2);        // [3072][1024]
    f16* wprojT = (f16*)alloc((size_t)1024 * 1024 * 2);        // [1024][1024]
    f16* w1T    = (f16*)alloc((size_t)4096 * 1024 * 2);        // [4096][1024]
    f16* w2T    = (f16*)alloc((size_t)1024 * 4096 * 2);        // [1024][4096]
    f16* hln    = (f16*)alloc((size_t)M * DIMW * 2);           // reused ln1/ln2
    f16* qkvh   = (f16*)alloc((size_t)M * 3 * DIMW * 2);
    f16* oh     = (f16*)alloc((size_t)M * DIMW * 2);
    float* x1   = (float*)alloc((size_t)M * DIMW * 4);
    f16* h3     = (f16*)alloc((size_t)M * 4096 * 2);

    // 1) weight transpose+cast: [K][N] fp32 -> [N][K] f16.
    //    w_qkv: fold attention scale 1/sqrt(64)=0.125 into the Q columns (n < 1024).
    transpose_cast_kernel<<<dim3(3072 / 32, 1024 / 32), 256, 0, stream>>>(w_qkv,  wqkvT,  1024, 3072, 1024, 0.125f);
    transpose_cast_kernel<<<dim3(1024 / 32, 1024 / 32), 256, 0, stream>>>(w_proj, wprojT, 1024, 1024, 0, 1.0f);
    transpose_cast_kernel<<<dim3(4096 / 32, 1024 / 32), 256, 0, stream>>>(w1,     w1T,    1024, 4096, 0, 1.0f);
    transpose_cast_kernel<<<dim3(1024 / 32, 4096 / 32), 256, 0, stream>>>(w2,     w2T,    4096, 1024, 0, 1.0f);
    // 2) LN1
    ln_kernel<<<(int)M, 128, 0, stream>>>(x, g1, be1, hln);
    // 3) QKV GEMM
    gemm_kernel<true, false, false, false><<<dim3(3072 / 128, M / 128), 256, 0, stream>>>(
        hln, wqkvT, nullptr, nullptr, qkvh, (int)M, 3072, 1024);
    // 4) attention
    attn_kernel<<<BATCHW * 16 * (SEQW / 128), 256, 0, stream>>>(qkvh, oh);
    // 5) proj + bias + residual(x) -> x1 fp32
    gemm_kernel<false, true, false, true><<<dim3(1024 / 128, M / 128), 256, 0, stream>>>(
        oh, wprojT, b_proj, x, x1, (int)M, 1024, 1024);
    // 6) LN2
    ln_kernel<<<(int)M, 128, 0, stream>>>(x1, g2, be2, hln);
    // 7) MLP1 + bias + gelu -> f16
    gemm_kernel<true, true, true, false><<<dim3(4096 / 128, M / 128), 256, 0, stream>>>(
        hln, w1T, b1, nullptr, h3, (int)M, 4096, 1024);
    // 8) MLP2 + bias + residual(x1) -> out fp32
    gemm_kernel<false, true, false, true><<<dim3(1024 / 128, M / 128), 256, 0, stream>>>(
        h3, w2T, b2, x1, out, (int)M, 1024, 4096);
}
